// AVLnetBaselineModel_63780264345997
// MI455X (gfx1250) — matrix operand order repaired and checked by reading, not compiled, not run on hardware
//
#include <hip/hip_runtime.h>
#include <math.h>

// ---------------------------------------------------------------------------
// CDNA5 (gfx1250) WMMA types / helpers
// ---------------------------------------------------------------------------
typedef __attribute__((ext_vector_type(16))) __bf16 v16bf;
typedef __attribute__((ext_vector_type(8)))  float  v8f;

__device__ __forceinline__ unsigned short f2bf(float f) {
    unsigned u = __float_as_uint(f);
    u += 0x7FFFu + ((u >> 16) & 1u);   // round-to-nearest-even
    return (unsigned short)(u >> 16);
}
__device__ __forceinline__ float bf2f(unsigned short h) {
    return __uint_as_float(((unsigned)h) << 16);
}

#define WMMA_BF16(a, b, c) \
    __builtin_amdgcn_wmma_f32_16x16x32_bf16(false, (a), false, (b), (short)0, (c), false, false)

union FragBF { v16bf v; uint4 q[2]; unsigned short u[16]; };

__device__ __forceinline__ void pack8(FragBF& f, int at, float4 a, float4 b) {
    f.u[at + 0] = f2bf(a.x); f.u[at + 1] = f2bf(a.y);
    f.u[at + 2] = f2bf(a.z); f.u[at + 3] = f2bf(a.w);
    f.u[at + 4] = f2bf(b.x); f.u[at + 5] = f2bf(b.y);
    f.u[at + 6] = f2bf(b.z); f.u[at + 7] = f2bf(b.w);
}

// ---------------------------------------------------------------------------
// Weight conversion: f32 [OC][IC][KW]  ->  bf16 [OC][KW][IC]
// ---------------------------------------------------------------------------
__global__ void cvt_conv_w_k(const float* __restrict__ src, unsigned short* __restrict__ dst,
                             int OC, int IC, int KW, long total) {
    long idx = (long)blockIdx.x * blockDim.x + threadIdx.x;
    if (idx >= total) return;
    int kw = (int)(idx % KW);
    long t = idx / KW;
    int ic = (int)(t % IC);
    int oc = (int)(t / IC);
    dst[((long)oc * KW + kw) * IC + ic] = f2bf(src[idx]);
}

// ---------------------------------------------------------------------------
// conv1 + batchnorm affine + relu.  audio [32][40][2048] -> act1 bf16 [32][2048][128]
// ---------------------------------------------------------------------------
__global__ void conv1_k(const float* __restrict__ audio, const float* __restrict__ c1w,
                        const float* __restrict__ c1b, const float* __restrict__ bng,
                        const float* __restrict__ bnb, unsigned short* __restrict__ act1) {
    __shared__ float sx[40][128];
    int b  = blockIdx.y;
    int w0 = blockIdx.x * 128;
    int t  = threadIdx.x;
    float s    = bng[0] * (1.0f / sqrtf(1.0f + 1e-5f));
    float beta = bnb[0];
    for (int h = 0; h < 40; ++h)
        sx[h][t] = audio[((long)b * 40 + h) * 2048 + w0 + t] * s + beta;
    __syncthreads();
    long obase = ((long)b * 2048 + w0 + t) * 128;
    for (int oc = 0; oc < 128; ++oc) {
        float sum = c1b[oc];
        #pragma unroll 8
        for (int h = 0; h < 40; ++h) sum += sx[h][t] * c1w[oc * 40 + h];
        act1[obase + oc] = f2bf(fmaxf(sum, 0.0f));
    }
}

// ---------------------------------------------------------------------------
// Implicit-GEMM 1D conv via WMMA bf16.
//   X   : bf16 [B][Win][IC]   (channel-contiguous, 32B-aligned rows)
//   Wgt : bf16 [OC][KW][IC]
//   Y   : f32  [B][Win][OC]   (same-width conv, relu applied)
// One wave computes 16(oc) x 32(w): one A (weight) fragment feeds two B tiles.
// ---------------------------------------------------------------------------
__global__ __launch_bounds__(32)
void conv_wmma_k(const unsigned short* __restrict__ X, const unsigned short* __restrict__ Wgt,
                 const float* __restrict__ bias, float* __restrict__ Y,
                 int Win, int IC, int OC, int KW, int pad) {
    int lane = threadIdx.x & 31;
    int row  = lane & 15;
    int hi   = lane >> 4;
    int b    = blockIdx.z;
    int oc0  = blockIdx.y * 16;
    int w0   = blockIdx.x * 32;

    v8f acc0 = {}, acc1 = {};
    const unsigned short* wrow = Wgt + (long)(oc0 + row) * KW * IC;
    long xbase = (long)b * Win * IC;

    for (int kw = 0; kw < KW; ++kw) {
        int wi0 = w0 + row + kw - pad;
        int wi1 = wi0 + 16;
        bool inb0 = (wi0 >= 0) && (wi0 < Win);
        bool inb1 = (wi1 >= 0) && (wi1 < Win);
        const unsigned short* xc0 = X + xbase + (long)wi0 * IC;
        const unsigned short* xc1 = X + xbase + (long)wi1 * IC;
        const unsigned short* wkw = wrow + (long)kw * IC;
        __builtin_prefetch(wkw, 0, 1);
        for (int ic0 = 0; ic0 < IC; ic0 += 32) {
            FragBF af, b0, b1;
            // A (weights): u[0..7] <- wkw[ic0+hi*8 ..], u[8..15] <- wkw[ic0+hi*8+16 ..]
            af.q[0] = *(const uint4*)(wkw + ic0 + hi * 8);
            af.q[1] = *(const uint4*)(wkw + ic0 + hi * 8 + 16);
            // B (acts): u[0..7] <- ic0+hi*8 .., u[8..15] <- ic0+hi*8+16 ..
            int baseB = ic0 + hi * 8;
            if (inb0) {
                b0.q[0] = *(const uint4*)(xc0 + baseB);
                b0.q[1] = *(const uint4*)(xc0 + baseB + 16);
            } else { b0.q[0] = make_uint4(0,0,0,0); b0.q[1] = make_uint4(0,0,0,0); }
            if (inb1) {
                b1.q[0] = *(const uint4*)(xc1 + baseB);
                b1.q[1] = *(const uint4*)(xc1 + baseB + 16);
            } else { b1.q[0] = make_uint4(0,0,0,0); b1.q[1] = make_uint4(0,0,0,0); }
            acc0 = WMMA_BF16(af.v, b0.v, acc0);
            acc1 = WMMA_BF16(af.v, b1.v, acc1);
        }
    }
    #pragma unroll
    for (int r = 0; r < 8; ++r) {
        int oc = oc0 + r + hi * 8;
        float bv = bias[oc];
        int wA = w0 + row;
        int wB = w0 + 16 + row;
        Y[((long)b * Win + wA) * OC + oc] = fmaxf(acc0[r] + bv, 0.0f);
        Y[((long)b * Win + wB) * OC + oc] = fmaxf(acc1[r] + bv, 0.0f);
    }
}

// ---------------------------------------------------------------------------
// maxpool width 3, stride 2, pad 1:  f32 [B][Wc][OC] -> bf16 [B][Wc/2][OC]
// ---------------------------------------------------------------------------
__global__ void pool_k(const float* __restrict__ Y, unsigned short* __restrict__ O,
                       int Wc, int OC, long total) {
    long idx = (long)blockIdx.x * blockDim.x + threadIdx.x;
    if (idx >= total) return;
    int Wout = Wc >> 1;
    int oc = (int)(idx % OC);
    long t = idx / OC;
    int wp = (int)(t % Wout);
    int b  = (int)(t / Wout);
    int w  = 2 * wp;
    const float* base = Y + ((long)b * Wc) * OC + oc;
    float m = base[(long)w * OC];
    if (w - 1 >= 0) m = fmaxf(m, base[(long)(w - 1) * OC]);
    if (w + 1 < Wc) m = fmaxf(m, base[(long)(w + 1) * OC]);
    O[idx] = f2bf(m);
}

// ---------------------------------------------------------------------------
// masked mean over time: act5 bf16 [32][128][1024] -> pooled f32 [32][1024]
// ---------------------------------------------------------------------------
__global__ void amean_k(const unsigned short* __restrict__ act5, const int* __restrict__ nframes,
                        float* __restrict__ pooled) {
    long idx = (long)blockIdx.x * blockDim.x + threadIdx.x;
    if (idx >= 32 * 1024) return;
    int c = (int)(idx & 1023);
    int b = (int)(idx >> 10);
    int nf = nframes[b] / 16;
    if (nf < 1) nf = 1;
    if (nf > 128) nf = 128;
    float s = 0.f;
    for (int t = 0; t < nf; ++t) s += bf2f(act5[((long)b * 128 + t) * 1024 + c]);
    pooled[idx] = s / (float)nf;
}

// ---------------------------------------------------------------------------
// GEMM via WMMA bf16:  C[M][N] = A[M][K] @ Wt[N][K]^T + bias, optional relu.
// f32 operands converted in-register.  Each wave computes a 32(m) x 16(n)
// tile (two m-tiles share one weight fragment -> weights streamed once).
// Requires M % 32 == 0.  Rows of A/Wt must be 16B aligned (K*4 % 16 == 0).
// ---------------------------------------------------------------------------
__global__ __launch_bounds__(128)
void gemm_wmma_k(const float* __restrict__ A, const float* __restrict__ Wt,
                 const float* __restrict__ bias, float* __restrict__ C,
                 int M, int N, int K, int relu) {
    int lane = threadIdx.x & 31;
    int wave = threadIdx.x >> 5;
    int n0 = (blockIdx.x * 4 + wave) * 16;
    int mbase = blockIdx.y * 32;
    int row = lane & 15;
    int hi  = lane >> 4;

    const float* a0   = A  + (long)(mbase + row) * K;
    const float* a1   = A  + (long)(mbase + 16 + row) * K;
    const float* wrow = Wt + (long)(n0 + row) * K;
    v8f acc0 = {}, acc1 = {};

    int Kv = K & ~31;                       // full 32-wide chunks
    for (int k0 = 0; k0 < Kv; k0 += 32) {
        __builtin_prefetch(wrow + k0 + 256, 0, 1);
        FragBF af0, af1, bfv;
        {   // A fragments: 16 consecutive at k0 + hi*16
            const float4* p = (const float4*)(a0 + k0 + hi * 16);
            const float4* q = (const float4*)(a0 + k0 + hi * 16 + 8);
            pack8(af0, 0, p[0], p[1]);
            pack8(af0, 8, q[0], q[1]);
            p = (const float4*)(a1 + k0 + hi * 16);
            q = (const float4*)(a1 + k0 + hi * 16 + 8);
            pack8(af1, 0, p[0], p[1]);
            pack8(af1, 8, q[0], q[1]);
        }
        {   // B fragment: 16 consecutive at k0 + hi*16
            const float4* p = (const float4*)(wrow + k0 + hi * 16);
            pack8(bfv, 0, p[0], p[1]);
            pack8(bfv, 8, p[2], p[3]);
        }
        acc0 = WMMA_BF16(af0.v, bfv.v, acc0);
        acc1 = WMMA_BF16(af1.v, bfv.v, acc1);
    }
    if (Kv < K) {                           // guarded scalar tail (K % 32 != 0)
        FragBF af0, af1, bfv;
        int baseA = Kv + hi * 16;
        #pragma unroll
        for (int e = 0; e < 8; ++e) {
            int k1 = baseA + e, k2 = baseA + 8 + e;
            af0.u[e]     = (k1 < K) ? f2bf(a0[k1]) : (unsigned short)0;
            af0.u[e + 8] = (k2 < K) ? f2bf(a0[k2]) : (unsigned short)0;
            af1.u[e]     = (k1 < K) ? f2bf(a1[k1]) : (unsigned short)0;
            af1.u[e + 8] = (k2 < K) ? f2bf(a1[k2]) : (unsigned short)0;
        }
        int baseB = Kv + hi * 16;
        #pragma unroll
        for (int e = 0; e < 16; ++e) {
            int k1 = baseB + e;
            bfv.u[e] = (k1 < K) ? f2bf(wrow[k1]) : (unsigned short)0;
        }
        acc0 = WMMA_BF16(af0.v, bfv.v, acc0);
        acc1 = WMMA_BF16(af1.v, bfv.v, acc1);
    }

    int n = n0 + row;
    float bv = bias[n];
    #pragma unroll
    for (int r = 0; r < 8; ++r) {
        int mA = mbase + r + hi * 8;
        int mB = mA + 16;
        float vA = acc0[r] + bv;
        float vB = acc1[r] + bv;
        if (relu) { vA = fmaxf(vA, 0.0f); vB = fmaxf(vB, 0.0f); }
        C[(long)mA * N + n] = vA;
        C[(long)mB * N + n] = vB;
    }
}

// ---------------------------------------------------------------------------
// rows-group max: X [B*G][N] -> O [B][N]
// ---------------------------------------------------------------------------
__global__ void rowmax_k(const float* __restrict__ X, float* __restrict__ O,
                         int G, int N, long total) {
    long idx = (long)blockIdx.x * blockDim.x + threadIdx.x;
    if (idx >= total) return;
    int c = (int)(idx % N);
    long b = idx / N;
    const float* p = X + (b * G) * (long)N + c;
    float m = p[0];
    for (int j = 1; j < G; ++j) m = fmaxf(m, p[(long)j * N]);
    O[idx] = m;
}

// ---------------------------------------------------------------------------
// gated-unit + L2 normalize per row (block = one row, 256 threads, N <= 4096)
// ---------------------------------------------------------------------------
__global__ __launch_bounds__(256)
void gate_norm_k(const float* __restrict__ x1, const float* __restrict__ x2,
                 float* __restrict__ out, int N, int gate) {
    __shared__ float red[256];
    int rowi = blockIdx.x;
    int t = threadIdx.x;
    const float* p1 = x1 + (long)rowi * N;
    const float* p2 = x2 + (long)rowi * N;
    int cnt = N / 256;
    float y[16];
    float ss = 0.f;
    for (int i = 0; i < cnt; ++i) {
        int c = t + i * 256;
        float v = p1[c];
        if (gate) {
            float g = p2[c];
            v = v * (1.0f / (1.0f + __expf(-g)));
        }
        y[i] = v;
        ss += v * v;
    }
    red[t] = ss;
    __syncthreads();
    for (int s = 128; s > 0; s >>= 1) {
        if (t < s) red[t] += red[t + s];
        __syncthreads();
    }
    float inv = 1.0f / fmaxf(sqrtf(red[0]), 1e-12f);
    for (int i = 0; i < cnt; ++i) out[(long)rowi * N + t + i * 256] = y[i] * inv;
}

// ---------------------------------------------------------------------------
// host side
// ---------------------------------------------------------------------------
static inline size_t alignup(size_t x) { return (x + 255) & ~(size_t)255; }

extern "C" void kernel_launch(void* const* d_in, const int* in_sizes, int n_in,
                              void* d_out, int out_size, void* d_ws, size_t ws_size,
                              hipStream_t stream) {
    (void)in_sizes; (void)n_in; (void)out_size; (void)ws_size;
    const float* text   = (const float*)d_in[0];
    const float* video  = (const float*)d_in[1];
    const float* audio  = (const float*)d_in[2];
    const int*   nfr    = (const int*)  d_in[3];
    const float* bng    = (const float*)d_in[4];
    const float* bnb    = (const float*)d_in[5];
    const float* c1w    = (const float*)d_in[6];   const float* c1b = (const float*)d_in[7];
    const float* c2w    = (const float*)d_in[8];   const float* c2b = (const float*)d_in[9];
    const float* c3w    = (const float*)d_in[10];  const float* c3b = (const float*)d_in[11];
    const float* c4w    = (const float*)d_in[12];  const float* c4b = (const float*)d_in[13];
    const float* c5w    = (const float*)d_in[14];  const float* c5b = (const float*)d_in[15];
    const float* projw  = (const float*)d_in[16];  const float* projb = (const float*)d_in[17];
    const float* guafw  = (const float*)d_in[18];  const float* guafb = (const float*)d_in[19];
    const float* guacw  = (const float*)d_in[20];  const float* guacb = (const float*)d_in[21];
    const float* guvfw  = (const float*)d_in[22];  const float* guvfb = (const float*)d_in[23];
    const float* guvcw  = (const float*)d_in[24];  const float* guvcb = (const float*)d_in[25];
    const float* tpw    = (const float*)d_in[26];  const float* tpb   = (const float*)d_in[27];
    const float* gutfw  = (const float*)d_in[28];  const float* gutfb = (const float*)d_in[29];
    const float* gutcw  = (const float*)d_in[30];  const float* gutcb = (const float*)d_in[31];

    float* out = (float*)d_out;                 // [3][32][4096]
    const int B = 32;

    char* ws = (char*)d_ws;
    size_t off = 0;
    auto take = [&](size_t bytes) { char* p = ws + off; off += alignup(bytes); return p; };

    unsigned short* wc2  = (unsigned short*)take((size_t)256 * 11 * 128 * 2);
    unsigned short* wc3  = (unsigned short*)take((size_t)512 * 17 * 256 * 2);
    unsigned short* wc4  = (unsigned short*)take((size_t)512 * 17 * 512 * 2);
    unsigned short* wc5  = (unsigned short*)take((size_t)1024 * 17 * 512 * 2);
    unsigned short* act1 = (unsigned short*)take((size_t)B * 2048 * 128 * 2);
    float*          cbuf = (float*)         take((size_t)B * 2048 * 256 * 4);
    unsigned short* act2 = (unsigned short*)take((size_t)B * 1024 * 256 * 2);
    unsigned short* act3 = (unsigned short*)take((size_t)B * 512  * 512 * 2);
    unsigned short* act4 = (unsigned short*)take((size_t)B * 256  * 512 * 2);
    unsigned short* act5 = (unsigned short*)take((size_t)B * 128  * 1024 * 2);
    float*          pooledA = (float*)      take((size_t)B * 1024 * 4);
    float*          tmp_t   = (float*)      take((size_t)B * 30 * 4096 * 4);
    float*          tbuf    = (float*)      take((size_t)B * 4096 * 4);
    float*          vbuf    = (float*)      take((size_t)B * 4096 * 4);
    float*          x1b     = (float*)      take((size_t)B * 4096 * 4);
    float*          x2b     = (float*)      take((size_t)B * 4096 * 4);
    float*          gau     = (float*)      take((size_t)B * 1024 * 4);

    // conv weight conversion
    {
        long t2 = 256L * 128 * 11, t3 = 512L * 256 * 17, t4 = 512L * 512 * 17, t5 = 1024L * 512 * 17;
        cvt_conv_w_k<<<dim3((t2 + 255) / 256), 256, 0, stream>>>(c2w, wc2, 256, 128, 11, t2);
        cvt_conv_w_k<<<dim3((t3 + 255) / 256), 256, 0, stream>>>(c3w, wc3, 512, 256, 17, t3);
        cvt_conv_w_k<<<dim3((t4 + 255) / 256), 256, 0, stream>>>(c4w, wc4, 512, 512, 17, t4);
        cvt_conv_w_k<<<dim3((t5 + 255) / 256), 256, 0, stream>>>(c5w, wc5, 1024, 512, 17, t5);
    }

    // ---- text branch ----
    gemm_wmma_k<<<dim3(4096 / 64, (B * 30) / 32), 128, 0, stream>>>(
        text, tpw, tpb, tmp_t, B * 30, 4096, 300, 1);
    { long tot = (long)B * 4096;
      rowmax_k<<<dim3((tot + 255) / 256), 256, 0, stream>>>(tmp_t, tbuf, 30, 4096, tot); }
    gemm_wmma_k<<<dim3(4096 / 64, 1), 128, 0, stream>>>(tbuf, gutfw, gutfb, x1b, B, 4096, 4096, 0);
    gemm_wmma_k<<<dim3(4096 / 64, 1), 128, 0, stream>>>(tbuf, gutcw, gutcb, x2b, B, 4096, 4096, 0);
    gate_norm_k<<<dim3(B), 256, 0, stream>>>(x1b, x2b, out + 0, 4096, 1);

    // ---- video branch ----
    { long tot = (long)B * 4096;
      rowmax_k<<<dim3((tot + 255) / 256), 256, 0, stream>>>(video, vbuf, 16, 4096, tot); }
    gate_norm_k<<<dim3(B), 256, 0, stream>>>(vbuf, vbuf, vbuf, 4096, 0);
    gemm_wmma_k<<<dim3(4096 / 64, 1), 128, 0, stream>>>(vbuf, guvfw, guvfb, x1b, B, 4096, 4096, 0);
    gemm_wmma_k<<<dim3(4096 / 64, 1), 128, 0, stream>>>(vbuf, guvcw, guvcb, x2b, B, 4096, 4096, 0);
    gate_norm_k<<<dim3(B), 256, 0, stream>>>(x1b, x2b, out + (size_t)B * 4096, 4096, 1);

    // ---- audio branch ----
    conv1_k<<<dim3(2048 / 128, B), 128, 0, stream>>>(audio, c1w, c1b, bng, bnb, act1);

    conv_wmma_k<<<dim3(2048 / 32, 256 / 16, B), 32, 0, stream>>>(act1, wc2, c2b, cbuf, 2048, 128, 256, 11, 5);
    { long tot = (long)B * 1024 * 256;
      pool_k<<<dim3((tot + 255) / 256), 256, 0, stream>>>(cbuf, act2, 2048, 256, tot); }

    conv_wmma_k<<<dim3(1024 / 32, 512 / 16, B), 32, 0, stream>>>(act2, wc3, c3b, cbuf, 1024, 256, 512, 17, 8);
    { long tot = (long)B * 512 * 512;
      pool_k<<<dim3((tot + 255) / 256), 256, 0, stream>>>(cbuf, act3, 1024, 512, tot); }

    conv_wmma_k<<<dim3(512 / 32, 512 / 16, B), 32, 0, stream>>>(act3, wc4, c4b, cbuf, 512, 512, 512, 17, 8);
    { long tot = (long)B * 256 * 512;
      pool_k<<<dim3((tot + 255) / 256), 256, 0, stream>>>(cbuf, act4, 512, 512, tot); }

    conv_wmma_k<<<dim3(256 / 32, 1024 / 16, B), 32, 0, stream>>>(act4, wc5, c5b, cbuf, 256, 512, 1024, 17, 8);
    { long tot = (long)B * 128 * 1024;
      pool_k<<<dim3((tot + 255) / 256), 256, 0, stream>>>(cbuf, act5, 256, 1024, tot); }

    amean_k<<<dim3((32 * 1024 + 255) / 256), 256, 0, stream>>>(act5, nfr, pooledA);

    gemm_wmma_k<<<dim3(1024 / 64, 1), 128, 0, stream>>>(pooledA, guafw, guafb, x1b, B, 1024, 1024, 0);
    gemm_wmma_k<<<dim3(1024 / 64, 1), 128, 0, stream>>>(pooledA, guacw, guacb, x2b, B, 1024, 1024, 0);
    gate_norm_k<<<dim3(B), 256, 0, stream>>>(x1b, x2b, gau, 1024, 1);
    gemm_wmma_k<<<dim3(4096 / 64, 1), 128, 0, stream>>>(gau, projw, projb,
                                                        out + (size_t)2 * B * 4096, B, 4096, 1024, 0);
}
